// MaskMambaBlock_51977694216537
// MI455X (gfx1250) — matrix operand order repaired and checked by reading, not compiled, not run on hardware
//
#include <hip/hip_runtime.h>
#include <hip/hip_bf16.h>
#include <cstdint>

// ---------------- problem constants (from reference) ----------------
constexpr int Bb  = 4;
constexpr int Tt  = 2048;
constexpr int Dd  = 1024;   // d_model
constexpr int DIi = 1024;   // d_inner
constexpr int Ns  = 16;     // d_state
constexpr int Rr  = 64;     // dt_rank
constexpr int Kc  = 4;      // conv kernel
constexpr int Mtok = Bb * Tt;   // 8192 tokens

// ---------------- vector types for WMMA ----------------
typedef _Float16 h8   __attribute__((ext_vector_type(8)));
typedef _Float16 v16h __attribute__((ext_vector_type(16)));
typedef float    v8f  __attribute__((ext_vector_type(8)));

union FragH { v16h v; h8 h[2]; };

// Async global->LDS 16B copy (CDNA5 TDM-lite path, tracked by ASYNCcnt).
// LDS destination = low 32 bits of the generic pointer (as3 offset).
__device__ __forceinline__ void async_copy_b128(uint32_t lds_addr, const void* gaddr)
{
    asm volatile("global_load_async_to_lds_b128 %0, %1, off"
                 :: "v"(lds_addr), "v"(gaddr) : "memory");
}
__device__ __forceinline__ uint32_t lds_off(const void* p)
{
    return (uint32_t)(uintptr_t)p;
}

// =====================================================================
// Generic WMMA f16 GEMM:  C[M,N] = A[M,K] * W[N,K]^T   (f32 accumulate)
//  - block tile: 128M x 64N, K-step 32; 8 waves, each 16M x 64N strip
//  - double-buffered LDS fed by global_load_async_to_lds_b128:
//      issue tile k+1 -> buf^1, s_wait_asynccnt<=3 (own tile-k done, async
//      ops retire in order), barrier, 4 WMMAs from buf, barrier.
//  - fragment reads are 16B ds_read_b128 matching wave32 WMMA layouts
// =====================================================================
__global__ __launch_bounds__(256)
void gemm_f16_wmma(const _Float16* __restrict__ A,
                   const _Float16* __restrict__ W,
                   float* __restrict__ C,
                   int M, int N, int K)
{
    constexpr int MB = 128, NB = 64, KB = 32;
    __shared__ __align__(16) _Float16 As[2][MB][KB + 8];
    __shared__ __align__(16) _Float16 Bs[2][NB][KB + 8];

    const int tid  = threadIdx.x;
    const int wave = tid >> 5;
    const int lane = tid & 31;
    const int lm   = lane & 15;     // row/col within 16
    const int lh   = lane >> 4;     // lane half selects K sub-chunk
    const int mBase = blockIdx.y * MB;
    const int nBase = blockIdx.x * NB;

    // fixed per-thread staging assignments (A: 512 chunks / 256 thr = 2 each;
    // B: 256 chunks = 1 each); OOB B rows are clamped (WMMA is column-
    // separable, garbage only reaches columns the store guard drops).
    const int ar0 = tid >> 2,          ac = (tid & 3) * 8;
    const int ar1 = (tid + 256) >> 2;
    const int br  = tid >> 2;
    const int gnc = min(nBase + br, N - 1);
    const _Float16* gA0 = A + (size_t)(mBase + ar0) * K + ac;
    const _Float16* gA1 = A + (size_t)(mBase + ar1) * K + ac;
    const _Float16* gB  = W + (size_t)gnc * K + ac;

    uint32_t lA0[2], lA1[2], lB[2];
    #pragma unroll
    for (int b = 0; b < 2; ++b) {
        lA0[b] = lds_off(&As[b][ar0][ac]);
        lA1[b] = lds_off(&As[b][ar1][ac]);
        lB[b]  = lds_off(&Bs[b][br][ac]);
    }

    v8f zero = {};
    v8f acc[4] = {zero, zero, zero, zero};

    const int nk = K / KB;
    // prime the pipeline: tile 0 -> buf 0
    async_copy_b128(lA0[0], gA0);
    async_copy_b128(lA1[0], gA1);
    async_copy_b128(lB[0],  gB);

    for (int ki = 0; ki < nk; ++ki) {
        const int buf = ki & 1;
        if (ki + 1 < nk) {
            const int k1 = (ki + 1) * KB;
            async_copy_b128(lA0[buf ^ 1], gA0 + k1);
            async_copy_b128(lA1[buf ^ 1], gA1 + k1);
            async_copy_b128(lB[buf ^ 1],  gB + k1);
            __builtin_prefetch(gA0 + k1 + KB, 0, 1);   // global_prefetch_b8
            asm volatile("s_wait_asynccnt 3" ::: "memory");
        } else {
            asm volatile("s_wait_asynccnt 0" ::: "memory");
        }
        __syncthreads();   // all waves' tile-ki async writes visible

        FragH a;
        {
            int row = wave * 16 + lm;
            a.h[0] = *(const h8*)&As[buf][row][lh ? 8 : 0];
            a.h[1] = *(const h8*)&As[buf][row][lh ? 24 : 16];
        }
        #pragma unroll
        for (int f = 0; f < 4; ++f) {
            FragH b;
            int row = f * 16 + lm;
            b.h[0] = *(const h8*)&Bs[buf][row][lh * 8];
            b.h[1] = *(const h8*)&Bs[buf][row][16 + lh * 8];
            acc[f] = __builtin_amdgcn_wmma_f32_16x16x32_f16(
                /*neg_a=*/false, a.v, /*neg_b=*/false, b.v,
                /*c_mod=*/(short)0, acc[f],
                /*reuse_a=*/false, /*reuse_b=*/false);
        }
        __syncthreads();   // buf free for reuse by iteration ki+1's issue (tile ki+2)
    }

    // ---- store C: f32 16x16 layout — VGPR r: lanes0-15 M=r, lanes16-31 M=r+8 ----
    #pragma unroll
    for (int f = 0; f < 4; ++f) {
        int gc = nBase + f * 16 + lm;
        if (gc >= N) continue;
        #pragma unroll
        for (int r = 0; r < 8; ++r) {
            int gm = mBase + wave * 16 + (lh ? r + 8 : r);
            C[(size_t)gm * N + gc] = acc[f][r];
        }
    }
    (void)M;
}

// =====================================================================
// small helpers
// =====================================================================
__global__ void cvt_f32_to_f16_kernel(const float* __restrict__ s,
                                      _Float16* __restrict__ d, int n)
{
    int i = blockIdx.x * 256 + threadIdx.x;
    if (i < n) d[i] = (_Float16)s[i];
}

__global__ void aneg_kernel(const float* __restrict__ A_log,
                            float* __restrict__ aneg, int n)
{
    int i = blockIdx.x * 256 + threadIdx.x;
    if (i < n) aneg[i] = -__expf(A_log[i]);
}

// LayerNorm pass 1: per-token mean / rstd (reads coalesced along T)
__global__ void ln_stats_kernel(const float* __restrict__ x,
                                float* __restrict__ mu,
                                float* __restrict__ rstd)
{
    int idx = blockIdx.x * 256 + threadIdx.x;     // token id
    if (idx >= Mtok) return;
    int b = idx >> 11, t = idx & (Tt - 1);
    size_t base = (size_t)b * Dd * Tt + t;
    float s = 0.f, ss = 0.f;
    for (int d = 0; d < Dd; ++d) {
        float v = x[base + (size_t)d * Tt];
        s += v; ss += v * v;
    }
    float m = s * (1.f / Dd);
    float var = ss * (1.f / Dd) - m * m;
    mu[idx] = m;
    rstd[idx] = rsqrtf(var + 1e-5f);
}

// LayerNorm pass 2: transpose (B,D,T)->(M,D) through LDS, normalize,
// emit both f32 (for later reuse) and f16 (GEMM operand).
__global__ __launch_bounds__(256)
void ln_apply_kernel(const float* __restrict__ x,
                     const float* __restrict__ mu,
                     const float* __restrict__ rstd,
                     const float* __restrict__ g,
                     const float* __restrict__ bln,
                     float* __restrict__ xn,
                     _Float16* __restrict__ xnh)
{
    __shared__ float tile[32][33];
    int b = blockIdx.z, c0 = blockIdx.y * 32, t0 = blockIdx.x * 32;
    int tid = threadIdx.x;
    #pragma unroll
    for (int i = 0; i < 4; ++i) {                 // read coalesced along T
        int dd = i * 8 + (tid >> 5), tc = tid & 31;
        tile[dd][tc] = x[(size_t)b * Dd * Tt + (size_t)(c0 + dd) * Tt + t0 + tc];
    }
    __syncthreads();
    #pragma unroll
    for (int i = 0; i < 4; ++i) {                 // write coalesced along D
        int tw = i * 8 + (tid >> 5), cw = tid & 31;
        int tok = b * Tt + t0 + tw;
        float v = (tile[cw][tw] - mu[tok]) * rstd[tok] * g[c0 + cw] + bln[c0 + cw];
        size_t o = (size_t)tok * Dd + c0 + cw;
        xn[o]  = v;
        xnh[o] = (_Float16)v;
    }
}

// phi[b,c] = relu(fc_w*mean_t(xn) + fc_b); reads token-major xn coalesced
__global__ void phi_kernel(const float* __restrict__ xn,
                           const float* __restrict__ fcw,
                           const float* __restrict__ fcb,
                           float* __restrict__ phi)
{
    int idx = blockIdx.x * 256 + threadIdx.x;     // b*Dd + d
    if (idx >= Bb * Dd) return;
    int b = idx >> 10, d = idx & (Dd - 1);
    float s = 0.f;
    for (int t = 0; t < Tt; ++t)
        s += xn[(size_t)(b * Tt + t) * Dd + d];
    float m = s * (1.f / Tt);
    phi[idx] = fmaxf(fcw[d] * m + fcb[d], 0.f);
}

// causal depthwise conv (K=4) + SiLU, emits f16 xm for x_proj GEMM
__global__ void conv_silu_kernel(const float* __restrict__ xz,
                                 const float* __restrict__ cw,
                                 const float* __restrict__ cb,
                                 _Float16* __restrict__ xmh)
{
    int i = blockIdx.x * 256 + threadIdx.x;       // m*DIi + d
    if (i >= Mtok * DIi) return;
    int m = i >> 10, d = i & (DIi - 1);
    int b = m >> 11, t = m & (Tt - 1);
    float acc = cb[d];
    #pragma unroll
    for (int k = 0; k < Kc; ++k) {
        int tt = t - (Kc - 1) + k;
        if (tt >= 0)
            acc += cw[d * Kc + k] * xz[(size_t)(b * Tt + tt) * (2 * DIi) + d];
    }
    float s = acc / (1.f + __expf(-acc));         // SiLU
    xmh[i] = (_Float16)s;
}

// slice x_dbl[:, :R] -> f16 A operand for dt GEMM
__global__ void dt_slice_kernel(const float* __restrict__ xdbl,
                                _Float16* __restrict__ dtA)
{
    int i = blockIdx.x * 256 + threadIdx.x;       // m*Rr + r
    if (i >= Mtok * Rr) return;
    dtA[i] = (_Float16)xdbl[(size_t)(i >> 6) * (Rr + 2 * Ns) + (i & (Rr - 1))];
}

// dt = softplus(dt_raw + dt_b), in place
__global__ void dt_act_kernel(float* __restrict__ dt,
                              const float* __restrict__ dtb)
{
    int i = blockIdx.x * 256 + threadIdx.x;
    if (i >= Mtok * DIi) return;
    float v = dt[i] + dtb[i & (DIi - 1)];
    dt[i] = (v > 20.f) ? v : log1pf(__expf(v));
}

// selective scan, fused with D-skip and SiLU(z) gating.
// one thread per (b, channel); B/C vectors (shared across channels) staged in LDS.
__global__ __launch_bounds__(256)
void scan_kernel(const float* __restrict__ xdbl,
                 const float* __restrict__ dt,
                 const _Float16* __restrict__ xmh,
                 const float* __restrict__ xz,
                 const float* __restrict__ aneg,
                 const float* __restrict__ Dp,
                 _Float16* __restrict__ yh)
{
    constexpr int TC = 64;
    __shared__ float sBC[TC][2 * Ns];
    int tid = threadIdx.x;
    int d = (blockIdx.x & 3) * 256 + tid;
    int b = blockIdx.x >> 2;

    float h[Ns], an[Ns];
    #pragma unroll
    for (int n = 0; n < Ns; ++n) { h[n] = 0.f; an[n] = aneg[d * Ns + n]; }
    float Dv = Dp[d];

    for (int t0 = 0; t0 < Tt; t0 += TC) {
        __syncthreads();
        for (int i = tid; i < TC * 2 * Ns; i += 256) {
            int tl = i >> 5, c = i & 31;
            sBC[tl][c] = xdbl[(size_t)(b * Tt + t0 + tl) * (Rr + 2 * Ns) + Rr + c];
        }
        __syncthreads();
        for (int tl = 0; tl < TC; ++tl) {
            size_t m = (size_t)(b * Tt + t0 + tl);
            float dtv = dt[m * DIi + d];
            float xmv = (float)xmh[m * DIi + d];
            float zv  = xz[m * (2 * DIi) + DIi + d];
            float dbx = dtv * xmv;
            float y = 0.f;
            #pragma unroll
            for (int n = 0; n < Ns; ++n) {
                float dA = __expf(dtv * an[n]);
                h[n] = dA * h[n] + dbx * sBC[tl][n];
                y += h[n] * sBC[tl][Ns + n];
            }
            y += Dv * xmv;
            y *= zv / (1.f + __expf(-zv));        // * silu(z)
            yh[m * DIi + d] = (_Float16)y;
        }
    }
}

// combine: xo(B,C,T) = x + dp_scale*((gf*phi + xn)*mask); LDS transpose back
__global__ __launch_bounds__(256)
void combine_kernel(const float* __restrict__ x,
                    const float* __restrict__ gf,     // (M,D)
                    const float* __restrict__ xn,     // (M,D)
                    const float* __restrict__ phi,    // (B,D)
                    const float* __restrict__ dp,
                    const unsigned char* __restrict__ mask,
                    float* __restrict__ out)
{
    __shared__ float tg[32][33], tx[32][33];
    int b = blockIdx.z, c0 = blockIdx.y * 32, t0 = blockIdx.x * 32;
    int tid = threadIdx.x;
    #pragma unroll
    for (int i = 0; i < 4; ++i) {                 // read coalesced along D
        int tt = i * 8 + (tid >> 5), cc = tid & 31;
        size_t m = (size_t)(b * Tt + t0 + tt) * Dd + c0 + cc;
        tg[tt][cc] = gf[m];
        tx[tt][cc] = xn[m];
    }
    __syncthreads();
    #pragma unroll
    for (int i = 0; i < 4; ++i) {                 // write coalesced along T
        int cw = i * 8 + (tid >> 5), tw = tid & 31;
        int c = c0 + cw, t = t0 + tw;
        float mk = mask[b * Tt + t] ? 1.f : 0.f;
        float o = (tg[tw][cw] * phi[b * Dd + c] + tx[tw][cw]) * mk;
        size_t xi = (size_t)b * Dd * Tt + (size_t)c * Tt + t;
        out[xi] = x[xi] + dp[c] * o;
    }
}

__global__ void mask_out_kernel(const unsigned char* __restrict__ mask,
                                float* __restrict__ out)
{
    int i = blockIdx.x * 256 + threadIdx.x;
    if (i < Bb * Tt) out[(size_t)Bb * Dd * Tt + i] = mask[i] ? 1.f : 0.f;
}

// =====================================================================
// launch
// =====================================================================
extern "C" void kernel_launch(void* const* d_in, const int* in_sizes, int n_in,
                              void* d_out, int out_size, void* d_ws, size_t ws_size,
                              hipStream_t stream)
{
    const float* x        = (const float*)d_in[0];
    const unsigned char* mask = (const unsigned char*)d_in[1];
    const float* ln_g     = (const float*)d_in[2];
    const float* ln_b     = (const float*)d_in[3];
    const float* fc_w     = (const float*)d_in[4];
    const float* fc_b     = (const float*)d_in[5];
    const float* dp_scale = (const float*)d_in[6];
    const float* in_proj_w  = (const float*)d_in[7];
    const float* conv_w   = (const float*)d_in[8];
    const float* conv_b   = (const float*)d_in[9];
    const float* x_proj_w = (const float*)d_in[10];
    const float* dt_w     = (const float*)d_in[11];
    const float* dt_b     = (const float*)d_in[12];
    const float* A_log    = (const float*)d_in[13];
    const float* D_param  = (const float*)d_in[14];
    const float* out_proj_w = (const float*)d_in[15];
    float* out = (float*)d_out;
    (void)in_sizes; (void)n_in; (void)out_size; (void)ws_size;

    uint8_t* w = (uint8_t*)d_ws;
    size_t off = 0;
    auto alloc = [&](size_t bytes) { void* p = w + off; off += (bytes + 255) & ~size_t(255); return p; };

    float*    xn_f32  = (float*)   alloc((size_t)Mtok * Dd * 4);
    _Float16* xn_h    = (_Float16*)alloc((size_t)Mtok * Dd * 2);
    _Float16* w_in_h  = (_Float16*)alloc((size_t)2 * DIi * Dd * 2);
    _Float16* w_xp_h  = (_Float16*)alloc((size_t)(Rr + 2 * Ns) * DIi * 2);
    _Float16* w_dt_h  = (_Float16*)alloc((size_t)DIi * Rr * 2);
    _Float16* w_out_h = (_Float16*)alloc((size_t)Dd * DIi * 2);
    float*    aneg    = (float*)   alloc((size_t)DIi * Ns * 4);
    float*    mu      = (float*)   alloc((size_t)Mtok * 4);
    float*    rstd    = (float*)   alloc((size_t)Mtok * 4);
    float*    phi     = (float*)   alloc((size_t)Bb * Dd * 4);
    float*    xz      = (float*)   alloc((size_t)Mtok * 2 * DIi * 4);
    _Float16* xm_h    = (_Float16*)alloc((size_t)Mtok * DIi * 2);
    float*    xdbl    = (float*)   alloc((size_t)Mtok * (Rr + 2 * Ns) * 4);
    _Float16* dtA_h   = (_Float16*)alloc((size_t)Mtok * Rr * 2);
    float*    dt      = (float*)   alloc((size_t)Mtok * DIi * 4);
    _Float16* y_h     = (_Float16*)alloc((size_t)Mtok * DIi * 2);
    float*    gf      = (float*)   alloc((size_t)Mtok * Dd * 4);

    auto blocks = [](int n) { return (n + 255) / 256; };

    // ---- weight conversions + A precompute ----
    cvt_f32_to_f16_kernel<<<blocks(2 * DIi * Dd), 256, 0, stream>>>(in_proj_w, w_in_h, 2 * DIi * Dd);
    cvt_f32_to_f16_kernel<<<blocks((Rr + 2 * Ns) * DIi), 256, 0, stream>>>(x_proj_w, w_xp_h, (Rr + 2 * Ns) * DIi);
    cvt_f32_to_f16_kernel<<<blocks(DIi * Rr), 256, 0, stream>>>(dt_w, w_dt_h, DIi * Rr);
    cvt_f32_to_f16_kernel<<<blocks(Dd * DIi), 256, 0, stream>>>(out_proj_w, w_out_h, Dd * DIi);
    aneg_kernel<<<blocks(DIi * Ns), 256, 0, stream>>>(A_log, aneg, DIi * Ns);

    // ---- LayerNorm (stats + transposed apply) ----
    ln_stats_kernel<<<blocks(Mtok), 256, 0, stream>>>(x, mu, rstd);
    ln_apply_kernel<<<dim3(Tt / 32, Dd / 32, Bb), 256, 0, stream>>>(x, mu, rstd, ln_g, ln_b, xn_f32, xn_h);
    phi_kernel<<<blocks(Bb * Dd), 256, 0, stream>>>(xn_f32, fc_w, fc_b, phi);

    // ---- in_proj GEMM: (M,1024) x (2048,1024)^T ----
    gemm_f16_wmma<<<dim3((2 * DIi) / 64, Mtok / 128), 256, 0, stream>>>(xn_h, w_in_h, xz, Mtok, 2 * DIi, Dd);

    // ---- causal depthwise conv + SiLU ----
    conv_silu_kernel<<<blocks(Mtok * DIi), 256, 0, stream>>>(xz, conv_w, conv_b, xm_h);

    // ---- x_proj GEMM: (M,1024) x (96,1024)^T ----
    gemm_f16_wmma<<<dim3((Rr + 2 * Ns + 63) / 64, Mtok / 128), 256, 0, stream>>>(xm_h, w_xp_h, xdbl, Mtok, Rr + 2 * Ns, DIi);

    // ---- dt_proj GEMM: (M,64) x (1024,64)^T, then softplus ----
    dt_slice_kernel<<<blocks(Mtok * Rr), 256, 0, stream>>>(xdbl, dtA_h);
    gemm_f16_wmma<<<dim3(DIi / 64, Mtok / 128), 256, 0, stream>>>(dtA_h, w_dt_h, dt, Mtok, DIi, Rr);
    dt_act_kernel<<<blocks(Mtok * DIi), 256, 0, stream>>>(dt, dt_b);

    // ---- selective scan (fused D-skip + silu(z) gate) ----
    scan_kernel<<<Bb * (DIi / 256), 256, 0, stream>>>(xdbl, dt, xm_h, xz, aneg, D_param, y_h);

    // ---- out_proj GEMM: (M,1024) x (1024,1024)^T ----
    gemm_f16_wmma<<<dim3(Dd / 64, Mtok / 128), 256, 0, stream>>>(y_h, w_out_h, gf, Mtok, Dd, DIi);

    // ---- combine + residual + mask; emit tuple outputs ----
    combine_kernel<<<dim3(Tt / 32, Dd / 32, Bb), 256, 0, stream>>>(x, gf, xn_f32, phi, dp_scale, mask, out);
    mask_out_kernel<<<blocks(Bb * Tt), 256, 0, stream>>>(mask, out);
}
